// SRNN_35064113004649
// MI455X (gfx1250) — compile-verified, matrix operand order checked by reading, not run on hardware
//
#include <hip/hip_runtime.h>
#include <stdint.h>

// ---------- constants (match reference) ----------
constexpr int Bc = 64, Tc = 256, Oc = 128, Xc = 512, Hc = 1024, Zc = 128;
constexpr int H3 = 3 * Hc;
#define LOG2PI 1.8378770664093453f

// ---------- WMMA types ----------
typedef __attribute__((ext_vector_type(16))) __bf16 bf16x16;
typedef __attribute__((ext_vector_type(8)))  float  f32x8;

__device__ __forceinline__ f32x8 zero8() {
  f32x8 z = {0.f,0.f,0.f,0.f,0.f,0.f,0.f,0.f};
  return z;
}

// Load a 16x32 bf16 fragment from a row-major [rows, ld] matrix.
// lane 0-15: row = base+lane, K-half 0; lane 16-31: row = base+lane-16, K-half 1.
// Each lane reads 16 contiguous bf16 (32B) — contiguous k order (k = 16h + s), same for A and B.
__device__ __forceinline__ bf16x16 ldfrag(const __bf16* M, int ld, int r0, int k, int lane) {
  int row = r0 + (lane & 15);
  int kk  = k + ((lane >> 4) << 4);
  return *reinterpret_cast<const bf16x16*>(M + row * ld + kk);
}
// Two-source variant: K in [0,K1) comes from A1, K in [K1,..) from A2 (concat inputs).
__device__ __forceinline__ bf16x16 ldfrag2(const __bf16* A1, int ld1, int K1,
                                           const __bf16* A2, int ld2,
                                           int r0, int k, int lane) {
  int row = r0 + (lane & 15);
  int kk  = k + ((lane >> 4) << 4);
  const __bf16* p = (kk < K1) ? (A1 + row * ld1 + kk)
                              : (A2 + row * ld2 + (kk - K1));
  return *reinterpret_cast<const bf16x16*>(p);
}

__device__ __forceinline__ float sigmoidf_(float x) { return 1.f / (1.f + expf(-x)); }
__device__ __forceinline__ float softplusf_(float x) { return (x > 20.f) ? x : log1pf(expf(x)); }

// flat pointer to a __shared__ object -> 32-bit LDS byte address
// (addrspace(3)->flat cast places the LDS offset in the low dword)
__device__ __forceinline__ unsigned lds_addr32(const void* p) {
  return (unsigned)(uintptr_t)p;
}
// Async per-lane 16B copy: global -> LDS, tracked by ASYNCcnt (CDNA5 TDM-lite path).
__device__ __forceinline__ void async_ld_b128(unsigned ldsOff, const void* gptr) {
  asm volatile("global_load_async_to_lds_b128 %0, %1, off"
               :: "v"(ldsOff), "v"(gptr) : "memory");
}
__device__ __forceinline__ void async_ld_b128_off16(unsigned ldsOff, const void* gptr) {
  asm volatile("global_load_async_to_lds_b128 %0, %1, off offset:16"
               :: "v"(ldsOff), "v"(gptr) : "memory");
}
__device__ __forceinline__ void wait_async0() {
  asm volatile("s_wait_asynccnt 0" ::: "memory");
}

// =====================================================================
// Generic LDS-tiled GEMM: out[M,N] = act(A[M,K] @ W[N,K]^T + bias)
// block 256 thr (8 waves, 4Mx2N), block tile 128x128, wave tile 32x64.
// Tiles staged memory->LDS with GLOBAL_LOAD_ASYNC_TO_LDS_B128 (ASYNCcnt).
// =====================================================================
__global__ void __launch_bounds__(256) gemm_bf16_128x128(
    const __bf16* __restrict__ A, const __bf16* __restrict__ W,
    const float* __restrict__ bias,
    float* __restrict__ outF, __bf16* __restrict__ outB,
    int M, int N, int K, int act)
{
  __shared__ __bf16 As[128][48];
  __shared__ __bf16 Ws[128][48];
  int tid = threadIdx.x, lane = tid & 31, wid = tid >> 5;
  int bm = blockIdx.y * 128, bn = blockIdx.x * 128;
  int mw = (wid & 3) * 32, nw = (wid >> 2) * 64;

  f32x8 c[2][4];
#pragma unroll
  for (int i = 0; i < 2; i++)
#pragma unroll
    for (int j = 0; j < 4; j++) c[i][j] = zero8();

  int srow = tid >> 1, scol = (tid & 1) * 16;
  unsigned ldsA = lds_addr32(&As[srow][scol]);
  unsigned ldsW = lds_addr32(&Ws[srow][scol]);

  for (int k = 0; k < K; k += 32) {
    // async-stage 128x32 tiles of A and W into LDS (2x16B per thread each)
    const __bf16* ga = A + (bm + srow) * K + k + scol;
    const __bf16* gw = W + (bn + srow) * K + k + scol;
    async_ld_b128(ldsA, ga);
    async_ld_b128_off16(ldsA, ga);
    async_ld_b128(ldsW, gw);
    async_ld_b128_off16(ldsW, gw);
    if (k + 32 < K) {
      __builtin_prefetch(A + (bm + srow) * K + k + 32 + scol, 0, 1);
      __builtin_prefetch(W + (bn + srow) * K + k + 32 + scol, 0, 1);
    }
    wait_async0();
    __syncthreads();

    bf16x16 af[2], wf[4];
#pragma unroll
    for (int mf = 0; mf < 2; mf++)
      af[mf] = *reinterpret_cast<const bf16x16*>(&As[mw + mf * 16 + (lane & 15)][(lane >> 4) * 16]);
#pragma unroll
    for (int nf = 0; nf < 4; nf++)
      wf[nf] = *reinterpret_cast<const bf16x16*>(&Ws[nw + nf * 16 + (lane & 15)][(lane >> 4) * 16]);
#pragma unroll
    for (int mf = 0; mf < 2; mf++)
#pragma unroll
      for (int nf = 0; nf < 4; nf++)
        c[mf][nf] = __builtin_amdgcn_wmma_f32_16x16x32_bf16(
            false, af[mf], false, wf[nf], (short)0, c[mf][nf], false, false);
    __syncthreads();
  }

  int nlo = lane & 15, mhi = (lane >> 4) * 8;
#pragma unroll
  for (int mf = 0; mf < 2; mf++)
#pragma unroll
    for (int nf = 0; nf < 4; nf++) {
      int col = bn + nw + nf * 16 + nlo;
      float bv = bias ? bias[col] : 0.f;
#pragma unroll
      for (int r = 0; r < 8; r++) {
        int m = bm + mw + mf * 16 + mhi + r;
        float v = c[mf][nf][r] + bv;
        if (act == 1) v = v > 0.f ? v : 0.f;
        if (outF) outF[(size_t)m * N + col] = v;
        if (outB) outB[(size_t)m * N + col] = (__bf16)v;
      }
    }
}

// =====================================================================
// Fused GRU step: hp = h_prev @ Whh^T (+bhh), gates, blend.
// grid 8 blocks x 256 thr; each wave owns 16 columns j of all 3 gates.
// =====================================================================
__global__ void __launch_bounds__(256) gru_step(
    const __bf16* __restrict__ hprev_bf, const float* __restrict__ hprev_f,
    const float* __restrict__ xproj, int useX,
    const __bf16* __restrict__ Whh,
    const float* __restrict__ bih, const float* __restrict__ bhh,
    float* __restrict__ hout_f, __bf16* __restrict__ hout_bf)
{
  int lane = threadIdx.x & 31, wid = threadIdx.x >> 5;
  int j0 = blockIdx.x * 128 + wid * 16;

  f32x8 c[3][4];
#pragma unroll
  for (int g = 0; g < 3; g++)
#pragma unroll
    for (int mf = 0; mf < 4; mf++) c[g][mf] = zero8();

  for (int k = 0; k < Hc; k += 32) {
    bf16x16 a[4];
#pragma unroll
    for (int mf = 0; mf < 4; mf++) a[mf] = ldfrag(hprev_bf, Hc, mf * 16, k, lane);
#pragma unroll
    for (int g = 0; g < 3; g++) {
      bf16x16 w = ldfrag(Whh, Hc, g * Hc + j0, k, lane);
#pragma unroll
      for (int mf = 0; mf < 4; mf++)
        c[g][mf] = __builtin_amdgcn_wmma_f32_16x16x32_bf16(
            false, a[mf], false, w, (short)0, c[g][mf], false, false);
    }
  }

  int nlo = lane & 15, mhi = (lane >> 4) * 8;
  int j = j0 + nlo;
  float br = bih[j], bz = bih[Hc + j], bn = bih[2 * Hc + j];
  float cr = bhh[j], cz = bhh[Hc + j], cn = bhh[2 * Hc + j];
#pragma unroll
  for (int mf = 0; mf < 4; mf++)
#pragma unroll
    for (int r = 0; r < 8; r++) {
      int m = mf * 16 + mhi + r;
      float xr = br, xz = bz, xn = bn;
      if (useX) {
        xr += xproj[m * H3 + j];
        xz += xproj[m * H3 + Hc + j];
        xn += xproj[m * H3 + 2 * Hc + j];
      }
      float hr = c[0][mf][r] + cr;
      float hz = c[1][mf][r] + cz;
      float hn = c[2][mf][r] + cn;
      float rg = sigmoidf_(xr + hr);
      float zg = sigmoidf_(xz + hz);
      float ng = tanhf(xn + rg * hn);
      float hv = (1.f - zg) * ng + zg * hprev_f[m * Hc + j];
      hout_f[m * Hc + j]  = hv;
      hout_bf[m * Hc + j] = (__bf16)hv;
    }
}

// =====================================================================
// Wave-level 64x64 tile GEMM (used by the fused Gaussian-head kernel)
// =====================================================================
__device__ void wave_tile64(const __bf16* A1, int ld1, int K1,
                            const __bf16* A2, int ld2,
                            const __bf16* W, int ldW, int n0, int K,
                            const float* bias, int act,
                            float* outF, int ldOF, __bf16* outB, int ldOB)
{
  int lane = threadIdx.x & 31;
  f32x8 c[4][4];
#pragma unroll
  for (int i = 0; i < 4; i++)
#pragma unroll
    for (int j = 0; j < 4; j++) c[i][j] = zero8();

  for (int k = 0; k < K; k += 32) {
    bf16x16 a[4];
#pragma unroll
    for (int mf = 0; mf < 4; mf++) a[mf] = ldfrag2(A1, ld1, K1, A2, ld2, mf * 16, k, lane);
#pragma unroll
    for (int nf = 0; nf < 4; nf++) {
      bf16x16 w = ldfrag(W, ldW, n0 + nf * 16, k, lane);
#pragma unroll
      for (int mf = 0; mf < 4; mf++)
        c[mf][nf] = __builtin_amdgcn_wmma_f32_16x16x32_bf16(
            false, a[mf], false, w, (short)0, c[mf][nf], false, false);
    }
  }

  int nlo = lane & 15, mhi = (lane >> 4) * 8;
#pragma unroll
  for (int mf = 0; mf < 4; mf++)
#pragma unroll
    for (int nf = 0; nf < 4; nf++) {
      int col = n0 + nf * 16 + nlo;
      float bv = bias ? bias[col] : 0.f;
#pragma unroll
      for (int r = 0; r < 8; r++) {
        int m = mf * 16 + mhi + r;
        float v = c[mf][nf][r] + bv;
        if (act == 1) v = v > 0.f ? v : 0.f;
        else if (act == 2) v = softplusf_(v) + 1e-6f;
        if (outF) outF[m * ldOF + col] = v;
        if (outB) outB[m * ldOB + col] = (__bf16)v;
      }
    }
}

// =====================================================================
// Fused Gaussian-head step: 3-layer relu MLP + mu / softplus-sd heads,
// optional reparameterized z-update. Single workgroup, 512 thr = 16 waves.
// Input = concat(main[64,Kmain], z[64,Zc]).
// =====================================================================
__global__ void __launch_bounds__(512) head_step(
    const __bf16* __restrict__ inMain, int ldMain, int Kmain,
    const __bf16* __restrict__ inZ,
    const __bf16* __restrict__ W1, const float* __restrict__ b1,
    const __bf16* __restrict__ W2, const float* __restrict__ b2,
    const __bf16* __restrict__ W3, const float* __restrict__ b3,
    const __bf16* __restrict__ Wmu, const float* __restrict__ bmu,
    const __bf16* __restrict__ Wsd, const float* __restrict__ bsd,
    __bf16* __restrict__ h1, __bf16* __restrict__ h2, __bf16* __restrict__ h3,
    float* __restrict__ muOut, float* __restrict__ sdOut,
    int doZ, const float* __restrict__ epsT, __bf16* __restrict__ zOut)
{
  int wid = threadIdx.x >> 5;
  int K0 = Kmain + Zc;
  // layer 1: [64,K0] -> relu -> h1[64,1024]
  wave_tile64(inMain, ldMain, Kmain, inZ, Zc, W1, K0, wid * 64, K0,
              b1, 1, nullptr, 0, h1, Hc);
  __threadfence(); __syncthreads();
  // layer 2
  wave_tile64(h1, Hc, Hc, h1, Hc, W2, Hc, wid * 64, Hc, b2, 1, nullptr, 0, h2, Hc);
  __threadfence(); __syncthreads();
  // layer 3
  wave_tile64(h2, Hc, Hc, h2, Hc, W3, Hc, wid * 64, Hc, b3, 1, nullptr, 0, h3, Hc);
  __threadfence(); __syncthreads();
  // heads: waves 0-1 -> mu, waves 2-3 -> softplus(sd)+1e-6
  if (wid < 2)
    wave_tile64(h3, Hc, Hc, h3, Hc, Wmu, Hc, wid * 64, Hc, bmu, 0, muOut, Zc, nullptr, 0);
  else if (wid < 4)
    wave_tile64(h3, Hc, Hc, h3, Hc, Wsd, Hc, (wid - 2) * 64, Hc, bsd, 2, sdOut, Zc, nullptr, 0);
  __threadfence(); __syncthreads();
  if (doZ) {
    for (int i = threadIdx.x; i < Bc * Zc; i += 512) {
      float zv = muOut[i] + sdOut[i] * epsT[i];
      zOut[i] = (__bf16)zv;
    }
  }
}

// ---------- small utility kernels ----------
__global__ void zero_f32(float* p, int n) {
  int i = blockIdx.x * blockDim.x + threadIdx.x;
  if (i < n) p[i] = 0.f;
}
__global__ void zero_bf16(__bf16* p, int n) {
  int i = blockIdx.x * blockDim.x + threadIdx.x;
  if (i < n) p[i] = (__bf16)0.f;
}
__global__ void f2bf(const float* __restrict__ s, __bf16* __restrict__ d, int n) {
  int i = blockIdx.x * blockDim.x + threadIdx.x;
  if (i < n) d[i] = (__bf16)s[i];
}
// x[B,T,O] -> xt_bf[(t*B+b)*O+o]
__global__ void build_xt(const float* __restrict__ x, __bf16* __restrict__ xt, int n) {
  int i = blockIdx.x * blockDim.x + threadIdx.x;
  if (i >= n) return;
  int o = i % Oc, b = (i / Oc) % Bc, t = i / (Oc * Bc);
  xt[i] = (__bf16)x[(b * Tc + t) * Oc + o];
}
// hx_rev[(t*B+b)*1536 + c] = concat(xe,d)[rev(t,b), b, c]
__global__ void build_hx(const __bf16* __restrict__ xe, const __bf16* __restrict__ dB,
                         const int* __restrict__ sl, __bf16* __restrict__ hx, int n) {
  int i = blockIdx.x * blockDim.x + threadIdx.x;
  if (i >= n) return;
  int c = i % (Xc + Hc), b = (i / (Xc + Hc)) % Bc, t = i / ((Xc + Hc) * Bc);
  int L = sl[b];
  int s = (t < L) ? (L - 1 - t) : t;
  hx[i] = (c < Xc) ? xe[(s * Bc + b) * Xc + c]
                   : dB[(size_t)s * Bc * Hc + b * Hc + (c - Xc)];
}
// a_bf[(t*B+b)*H + c] = hs_b[rev(t,b)] (slot s+1 of backward state buffer)
__global__ void build_a(const __bf16* __restrict__ hbB, const int* __restrict__ sl,
                        __bf16* __restrict__ a, int n) {
  int i = blockIdx.x * blockDim.x + threadIdx.x;
  if (i >= n) return;
  int c = i % Hc, b = (i / Hc) % Bc, t = i / (Hc * Bc);
  int L = sl[b];
  int s = (t < L) ? (L - 1 - t) : t;
  a[i] = hbB[(size_t)(s + 1) * Bc * Hc + b * Hc + c];
}
// dec_in[(t*B+b)*1152 + c] = concat(z_t, d_t)
__global__ void build_decin(const __bf16* __restrict__ zB, const __bf16* __restrict__ dB,
                            __bf16* __restrict__ di, int n) {
  int i = blockIdx.x * blockDim.x + threadIdx.x;
  if (i >= n) return;
  int c = i % (Zc + Hc), b = (i / (Zc + Hc)) % Bc, t = i / ((Zc + Hc) * Bc);
  di[i] = (c < Zc) ? zB[(size_t)(t + 1) * Bc * Zc + b * Zc + c]
                   : dB[(size_t)t * Bc * Hc + b * Hc + (c - Zc)];
}

// masked log-prob + KL row reduction; one block per (t,b) row
__global__ void __launch_bounds__(256) loss_rows(
    const float* __restrict__ x, const int* __restrict__ sl,
    const float* __restrict__ logits,
    const float* __restrict__ pm, const float* __restrict__ ps,
    const float* __restrict__ em, const float* __restrict__ es,
    float* __restrict__ accum)
{
  __shared__ float red[256];
  int row = blockIdx.x, tid = threadIdx.x;
  int t = row / Bc, b = row % Bc;
  float v = 0.f;
  if (t < sl[b]) {
    if (tid < Oc) {
      float mu  = logits[(size_t)row * (2 * Oc) + tid];
      float raw = logits[(size_t)row * (2 * Oc) + Oc + tid];
      float sd  = softplusf_(raw) + 1e-6f;
      float y   = x[(b * Tc + t) * Oc + tid];
      float dm  = (y - mu) / sd;
      v = -0.5f * (dm * dm + LOG2PI) - logf(sd);          // +lp
    } else {
      int zi = tid - Oc;
      size_t k = (size_t)row * Zc + zi;
      float P = ps[k], E = es[k], dM = em[k] - pm[k];
      float kld = logf(P / E) + (E * E + dM * dM) / (2.f * P * P) - 0.5f;
      v = -kld;                                            // -kl
    }
  }
  red[tid] = v;
  __syncthreads();
  for (int s = 128; s > 0; s >>= 1) {
    if (tid < s) red[tid] += red[tid + s];
    __syncthreads();
  }
  if (tid == 0) atomicAdd(accum, red[0]);
}

__global__ void finalize(const float* __restrict__ accum, const int* __restrict__ sl,
                         float* __restrict__ out) {
  if (threadIdx.x == 0) {
    int s = 0;
    for (int b = 0; b < Bc; b++) s += sl[b];
    out[0] = -accum[0] / (float)s;
  }
}

// =====================================================================
// host launch
// =====================================================================
static inline dim3 g1(int n) { return dim3((n + 255) / 256); }

extern "C" void kernel_launch(void* const* d_in, const int* in_sizes, int n_in,
                              void* d_out, int out_size, void* d_ws, size_t ws_size,
                              hipStream_t stream) {
  (void)in_sizes; (void)n_in; (void)out_size; (void)ws_size;
  // ---- inputs (setup_inputs order) ----
  const float* x      = (const float*)d_in[0];
  const int*   x_sl   = (const int*)d_in[1];
  const float* eps    = (const float*)d_in[2];
  const float* embW   = (const float*)d_in[3];
  const float* embB   = (const float*)d_in[4];
  const float* WihF   = (const float*)d_in[5];
  const float* WhhF   = (const float*)d_in[6];
  const float* bihF   = (const float*)d_in[7];
  const float* bhhF   = (const float*)d_in[8];
  const float* WihB   = (const float*)d_in[9];
  const float* WhhB   = (const float*)d_in[10];
  const float* bihB   = (const float*)d_in[11];
  const float* bhhB   = (const float*)d_in[12];
  const float* pW[5]  = { (const float*)d_in[13], (const float*)d_in[15], (const float*)d_in[17],
                          (const float*)d_in[19], (const float*)d_in[21] };
  const float* pB[5]  = { (const float*)d_in[14], (const float*)d_in[16], (const float*)d_in[18],
                          (const float*)d_in[20], (const float*)d_in[22] };
  const float* eW[5]  = { (const float*)d_in[23], (const float*)d_in[25], (const float*)d_in[27],
                          (const float*)d_in[29], (const float*)d_in[31] };
  const float* eB[5]  = { (const float*)d_in[24], (const float*)d_in[26], (const float*)d_in[28],
                          (const float*)d_in[30], (const float*)d_in[32] };
  const float* dW[3]  = { (const float*)d_in[33], (const float*)d_in[35], (const float*)d_in[37] };
  const float* dBs[3] = { (const float*)d_in[34], (const float*)d_in[36], (const float*)d_in[38] };
  const float* likW   = (const float*)d_in[39];
  const float* likB   = (const float*)d_in[40];
  float* out = (float*)d_out;

  const size_t MT = (size_t)Tc * Bc;      // 16384 rows
  // ---- workspace bump allocator ----
  char* p = (char*)d_ws;
  auto alloc = [&](size_t bytes) -> char* {
    char* r = (char*)(((uintptr_t)p + 255) & ~(uintptr_t)255);
    p = r + bytes;
    return r;
  };
  auto abf = [&](size_t n) { return (__bf16*)alloc(n * sizeof(__bf16)); };
  auto af  = [&](size_t n) { return (float*)alloc(n * sizeof(float)); };

  __bf16* xt_bf  = abf(MT * Oc);
  __bf16* xe_bf  = abf(MT * Xc);
  __bf16* wEmb   = abf((size_t)Xc * Oc);
  __bf16* wIhF   = abf((size_t)H3 * Xc);
  __bf16* wHhF   = abf((size_t)H3 * Hc);
  __bf16* wIhB   = abf((size_t)H3 * (Xc + Hc));
  __bf16* wHhB   = abf((size_t)H3 * Hc);
  __bf16* pWb[5], *eWb[5], *dWb[3];
  const size_t hwsz[5] = { (size_t)Hc * (Hc + Zc), (size_t)Hc * Hc, (size_t)Hc * Hc,
                           (size_t)Zc * Hc, (size_t)Zc * Hc };
  for (int i = 0; i < 5; i++) pWb[i] = abf(hwsz[i]);
  for (int i = 0; i < 5; i++) eWb[i] = abf(hwsz[i]);
  const size_t dwsz[3] = { (size_t)Hc * (Zc + Hc), (size_t)Hc * Hc, (size_t)Hc * Hc };
  for (int i = 0; i < 3; i++) dWb[i] = abf(dwsz[i]);
  __bf16* wLik   = abf((size_t)2 * Oc * Hc);

  float*  projF  = af(MT * H3);
  float*  projB  = af(MT * H3);
  float*  dF     = af((size_t)(Tc + 1) * Bc * Hc);
  __bf16* dB     = abf((size_t)(Tc + 1) * Bc * Hc);
  float*  hbF    = af((size_t)(Tc + 1) * Bc * Hc);
  __bf16* hbB    = abf((size_t)(Tc + 1) * Bc * Hc);
  __bf16* hx_bf  = abf(MT * (Xc + Hc));
  __bf16* a_bf   = abf(MT * Hc);
  __bf16* z_bf   = abf((size_t)(Tc + 1) * Bc * Zc);
  float*  pmF    = af(MT * Zc);
  float*  psF    = af(MT * Zc);
  float*  emF    = af(MT * Zc);
  float*  esF    = af(MT * Zc);
  __bf16* th1    = abf((size_t)Bc * Hc);
  __bf16* th2    = abf((size_t)Bc * Hc);
  __bf16* th3    = abf((size_t)Bc * Hc);
  __bf16* decin  = abf(MT * (Zc + Hc));
  __bf16* dh1    = abf(MT * Hc);
  __bf16* dh2    = abf(MT * Hc);
  __bf16* dh3    = abf(MT * Hc);
  float*  logits = af(MT * 2 * Oc);
  float*  accum  = af(1);

  // ---- weight conversion fp32 -> bf16 ----
  auto cvt = [&](const float* s, __bf16* d, size_t n) {
    f2bf<<<g1((int)n), 256, 0, stream>>>(s, d, (int)n);
  };
  cvt(embW, wEmb, (size_t)Xc * Oc);
  cvt(WihF, wIhF, (size_t)H3 * Xc);
  cvt(WhhF, wHhF, (size_t)H3 * Hc);
  cvt(WihB, wIhB, (size_t)H3 * (Xc + Hc));
  cvt(WhhB, wHhB, (size_t)H3 * Hc);
  for (int i = 0; i < 5; i++) cvt(pW[i], pWb[i], hwsz[i]);
  for (int i = 0; i < 5; i++) cvt(eW[i], eWb[i], hwsz[i]);
  for (int i = 0; i < 3; i++) cvt(dW[i], dWb[i], dwsz[i]);
  cvt(likW, wLik, (size_t)2 * Oc * Hc);

  // ---- zero initial states / accumulator ----
  zero_f32 <<<g1(Bc * Hc), 256, 0, stream>>>(dF, Bc * Hc);
  zero_bf16<<<g1(Bc * Hc), 256, 0, stream>>>(dB, Bc * Hc);
  zero_f32 <<<g1(Bc * Hc), 256, 0, stream>>>(hbF, Bc * Hc);
  zero_bf16<<<g1(Bc * Hc), 256, 0, stream>>>(hbB, Bc * Hc);
  zero_bf16<<<g1(Bc * Zc), 256, 0, stream>>>(z_bf, Bc * Zc);
  zero_f32 <<<1, 1, 0, stream>>>(accum, 1);

  // ---- embedding: xe = x @ embW^T + b   ([T,B] row order) ----
  build_xt<<<g1((int)(MT * Oc)), 256, 0, stream>>>(x, xt_bf, (int)(MT * Oc));
  gemm_bf16_128x128<<<dim3(Xc / 128, (int)(MT / 128)), 256, 0, stream>>>(
      xt_bf, wEmb, embB, nullptr, xe_bf, (int)MT, Xc, Oc, 0);

  // ---- forward GRU input projection (no bias; bias added in step) ----
  gemm_bf16_128x128<<<dim3(H3 / 128, (int)(MT / 128)), 256, 0, stream>>>(
      xe_bf, wIhF, nullptr, projF, nullptr, (int)MT, H3, Xc, 0);

  // ---- forward GRU scan: d slot t = h_{t-1}; step t writes slot t+1 ----
  for (int t = 0; t < Tc; t++) {
    const float* xp = (t == 0) ? nullptr : (projF + (size_t)(t - 1) * Bc * H3);
    gru_step<<<8, 256, 0, stream>>>(
        dB + (size_t)t * Bc * Hc, dF + (size_t)t * Bc * Hc, xp, t == 0 ? 0 : 1,
        wHhF, bihF, bhhF,
        dF + (size_t)(t + 1) * Bc * Hc, dB + (size_t)(t + 1) * Bc * Hc);
  }

  // ---- hx = reverse(concat(xe, d)); backward projection ----
  build_hx<<<g1((int)(MT * (Xc + Hc))), 256, 0, stream>>>(
      xe_bf, dB, x_sl, hx_bf, (int)(MT * (Xc + Hc)));
  gemm_bf16_128x128<<<dim3(H3 / 128, (int)(MT / 128)), 256, 0, stream>>>(
      hx_bf, wIhB, nullptr, projB, nullptr, (int)MT, H3, Xc + Hc, 0);

  // ---- backward GRU scan over reversed sequence ----
  for (int t = 0; t < Tc; t++) {
    gru_step<<<8, 256, 0, stream>>>(
        hbB + (size_t)t * Bc * Hc, hbF + (size_t)t * Bc * Hc,
        projB + (size_t)t * Bc * H3, 1,
        wHhB, bihB, bhhB,
        hbF + (size_t)(t + 1) * Bc * Hc, hbB + (size_t)(t + 1) * Bc * Hc);
  }
  build_a<<<g1((int)(MT * Hc)), 256, 0, stream>>>(hbB, x_sl, a_bf, (int)(MT * Hc));

  // ---- latent scan: prior head + encoder head + z reparam ----
  for (int t = 0; t < Tc; t++) {
    head_step<<<1, 512, 0, stream>>>(
        dB + (size_t)t * Bc * Hc, Hc, Hc, z_bf + (size_t)t * Bc * Zc,
        pWb[0], pB[0], pWb[1], pB[1], pWb[2], pB[2], pWb[3], pB[3], pWb[4], pB[4],
        th1, th2, th3,
        pmF + (size_t)t * Bc * Zc, psF + (size_t)t * Bc * Zc,
        0, nullptr, nullptr);
    head_step<<<1, 512, 0, stream>>>(
        a_bf + (size_t)t * Bc * Hc, Hc, Hc, z_bf + (size_t)t * Bc * Zc,
        eWb[0], eB[0], eWb[1], eB[1], eWb[2], eB[2], eWb[3], eB[3], eWb[4], eB[4],
        th1, th2, th3,
        emF + (size_t)t * Bc * Zc, esF + (size_t)t * Bc * Zc,
        1, eps + (size_t)t * Bc * Zc, z_bf + (size_t)(t + 1) * Bc * Zc);
  }

  // ---- decoder MLP + likelihood head ----
  build_decin<<<g1((int)(MT * (Zc + Hc))), 256, 0, stream>>>(
      z_bf, dB, decin, (int)(MT * (Zc + Hc)));
  gemm_bf16_128x128<<<dim3(Hc / 128, (int)(MT / 128)), 256, 0, stream>>>(
      decin, dWb[0], dBs[0], nullptr, dh1, (int)MT, Hc, Zc + Hc, 1);
  gemm_bf16_128x128<<<dim3(Hc / 128, (int)(MT / 128)), 256, 0, stream>>>(
      dh1, dWb[1], dBs[1], nullptr, dh2, (int)MT, Hc, Hc, 1);
  gemm_bf16_128x128<<<dim3(Hc / 128, (int)(MT / 128)), 256, 0, stream>>>(
      dh2, dWb[2], dBs[2], nullptr, dh3, (int)MT, Hc, Hc, 1);
  gemm_bf16_128x128<<<dim3((2 * Oc) / 128, (int)(MT / 128)), 256, 0, stream>>>(
      dh3, wLik, likB, logits, nullptr, (int)MT, 2 * Oc, Hc, 0);

  // ---- masked NLL + KL reduction, finalize ----
  loss_rows<<<(int)MT, 256, 0, stream>>>(x, x_sl, logits, pmF, psF, emF, esF, accum);
  finalize<<<1, 32, 0, stream>>>(accum, x_sl, out);
}
